// VQVAE_22179211117209
// MI455X (gfx1250) — matrix operand order repaired and checked by reading, not compiled, not run on hardware
//
#include <hip/hip_runtime.h>
#include <hip/hip_bf16.h>

// ---------------------------------------------------------------------------
// CDNA5 (gfx1250) types / helpers
// ---------------------------------------------------------------------------
typedef __attribute__((ext_vector_type(16))) __bf16 v16bf;
typedef __attribute__((ext_vector_type(8)))  float  v8f;

union FragBF { v16bf v; unsigned u[8]; };
union AccU   { v8f v;   float f[8];   };

__device__ __forceinline__ unsigned f2bf1(float f) {
    unsigned u = __float_as_uint(f);
    return (u + 0x7FFFu + ((u >> 16) & 1u)) >> 16;   // round-to-nearest-even
}
__device__ __forceinline__ unsigned packbf(float lo, float hi) {
    return f2bf1(lo) | (f2bf1(hi) << 16);
}
__device__ __forceinline__ float gelu_f(float x) {
    float x3 = x * x * x;
    return 0.5f * x * (1.f + tanhf(0.7978845608f * (x + 0.044715f * x3)));
}
__device__ __forceinline__ v8f wmma_bf16(const FragBF& a, const FragBF& b, v8f c) {
    return __builtin_amdgcn_wmma_f32_16x16x32_bf16(
        /*neg_a=*/false, a.v, /*neg_b=*/false, b.v,
        /*c_mod=*/(short)0, c, /*reuse_a=*/false, /*reuse_b=*/false);
}

#define FLAG_BIAS 1
#define FLAG_RES  2
#define FLAG_GELU 4

// ---------------------------------------------------------------------------
// Generic GEMM:  Y[M,N] = epilogue(A[M,K] @ W[K,N])
// 128x64 tile per workgroup, 8 waves each computing 32x32 via 4 WMMA tiles.
// A and W are fp32 in global; converted to bf16 while staging to LDS.
// ---------------------------------------------------------------------------
__global__ __launch_bounds__(256) void gemm_kernel(
        const float* __restrict__ A, const float* __restrict__ W,
        const float* __restrict__ bias, const float* __restrict__ res,
        float* __restrict__ Y, int M, int N, int K, int flags) {
    __shared__ unsigned As[128 * 16];   // 128 rows x 32 bf16 (16 dwords/row)
    __shared__ unsigned Bs[64 * 16];    // 64 cols  x 32 bf16 (transposed W)

    int t    = threadIdx.x;
    int m0   = blockIdx.y * 128, n0 = blockIdx.x * 64;
    int wave = t >> 5, lane = t & 31, hf = lane >> 4, l15 = lane & 15;
    int wr   = wave >> 1, wc = wave & 1;

    v8f zero = {0.f, 0.f, 0.f, 0.f, 0.f, 0.f, 0.f, 0.f};
    AccU acc[2][2];
    for (int mi = 0; mi < 2; mi++)
        for (int ni = 0; ni < 2; ni++) acc[mi][ni].v = zero;

    int ar = t >> 1, ah = t & 1;          // A-stage: row 0..127, half 0/1
    int bn = t & 63, bk = (t >> 6) * 8;   // B-stage: col 0..63,  k 0/8/16/24

    for (int kt = 0; kt < K; kt += 32) {
        // --- stage A tile (fp32 -> bf16) ---
        const float* Ap = A + (long)(m0 + ar) * K + kt + ah * 16;
        __builtin_prefetch(Ap + 32, 0, 1);
        unsigned* Ad = As + ar * 16 + ah * 8;
        #pragma unroll
        for (int j = 0; j < 16; j += 2) Ad[j >> 1] = packbf(Ap[j], Ap[j + 1]);
        // --- stage B tile transposed ---
        {
            const float* Wp = W + (long)(kt + bk) * N + n0 + bn;
            unsigned short* Bh = (unsigned short*)Bs;
            #pragma unroll
            for (int j = 0; j < 8; j++)
                Bh[bn * 32 + bk + j] = (unsigned short)f2bf1(Wp[(long)j * N]);
        }
        __syncthreads();

        FragBF afr[2], bfr[2];
        #pragma unroll
        for (int mi = 0; mi < 2; mi++) {
            int m = wr * 32 + mi * 16 + l15;
            const unsigned* p = As + m * 16 + hf * 4;
            #pragma unroll
            for (int j = 0; j < 4; j++) { afr[mi].u[j] = p[j]; afr[mi].u[4 + j] = p[8 + j]; }
        }
        #pragma unroll
        for (int ni = 0; ni < 2; ni++) {
            int n = wc * 32 + ni * 16 + l15;
            const unsigned* p = Bs + n * 16 + hf * 4;   // B layout: K=hf*8+i (i<8), 16+hf*8+(i-8)
            #pragma unroll
            for (int j = 0; j < 4; j++) { bfr[ni].u[j] = p[j]; bfr[ni].u[4 + j] = p[8 + j]; }
        }
        #pragma unroll
        for (int mi = 0; mi < 2; mi++)
            #pragma unroll
            for (int ni = 0; ni < 2; ni++)
                acc[mi][ni].v = wmma_bf16(afr[mi], bfr[ni], acc[mi][ni].v);
        __syncthreads();
    }

    // epilogue: bias / gelu / residual, fp32 store
    for (int mi = 0; mi < 2; mi++)
        for (int ni = 0; ni < 2; ni++) {
            int col = n0 + wc * 32 + ni * 16 + l15;
            #pragma unroll
            for (int vi = 0; vi < 8; vi++) {
                long row = m0 + wr * 32 + mi * 16 + vi + 8 * hf;
                float val = acc[mi][ni].f[vi];
                if (flags & FLAG_BIAS) val += bias[col];
                if (flags & FLAG_GELU) val = gelu_f(val);
                if (flags & FLAG_RES)  val += res[row * (long)N + col];
                Y[row * (long)N + col] = val;
            }
        }
}

// ---------------------------------------------------------------------------
// Windowed attention: one workgroup per (window, head).
// qkv row layout: [q(h*64) | k(512+h*64) | v(1024+h*64)], row stride 1536.
// LDS: Q,K bf16 (row major), V bf16 transposed, per-wave 16x256 f32 scores.
// ---------------------------------------------------------------------------
#define ATTN_SMEM (3 * 256 * 64 * 2 + 8 * 16 * 256 * 4)   // 229376 B < 320 KB

__global__ __launch_bounds__(256) void attn_kernel(
        const float* __restrict__ qkv, float* __restrict__ o) {
    extern __shared__ char smem[];
    unsigned*       Qw = (unsigned*)smem;           // 256 x 32 dwords
    unsigned*       Kw = Qw + 256 * 32;
    unsigned short* Vh = (unsigned short*)(Kw + 256 * 32);  // [64][256] bf16
    float*       Sbase = (float*)(smem + 3 * 256 * 64 * 2); // 8 x (16x256) f32

    int t   = threadIdx.x;
    int win = blockIdx.x >> 3, h = blockIdx.x & 7;
    long row0 = (long)win * 256;

    // stage: each thread converts one full row of q,k,v
    {
        const float* q = qkv + (row0 + t) * 1536 + h * 64;
        const float* k = q + 512;
        const float* v = q + 1024;
        unsigned* Qd = Qw + t * 32;
        unsigned* Kd = Kw + t * 32;
        #pragma unroll
        for (int c = 0; c < 64; c += 2) {
            Qd[c >> 1] = packbf(q[c], q[c + 1]);
            Kd[c >> 1] = packbf(k[c], k[c + 1]);
        }
        #pragma unroll
        for (int c = 0; c < 64; c++) Vh[c * 256 + t] = (unsigned short)f2bf1(v[c]);
    }
    __syncthreads();

    int wave = t >> 5, lane = t & 31, hf = lane >> 4, l15 = lane & 15;
    float* Sw = Sbase + wave * 16 * 256;
    const unsigned* Vw = (const unsigned*)Vh;   // dword view, row stride 128
    const float scale = 0.125f;                 // 64^-0.5
    v8f zero = {0.f, 0.f, 0.f, 0.f, 0.f, 0.f, 0.f, 0.f};

    for (int rbsel = 0; rbsel < 2; rbsel++) {
        int rb = wave * 2 + rbsel;              // 16-row block of this wave

        // Q fragments (A matrix), k = 0..63 in two 32-chunks
        FragBF aq[2];
        #pragma unroll
        for (int kk = 0; kk < 2; kk++) {
            int m = rb * 16 + l15;
            const unsigned* p = Qw + m * 32 + kk * 16 + hf * 4;
            #pragma unroll
            for (int j = 0; j < 4; j++) { aq[kk].u[j] = p[j]; aq[kk].u[4 + j] = p[8 + j]; }
        }
        // S = Q K^T * scale  -> per-wave f32 strip in LDS
        for (int cb = 0; cb < 16; cb++) {
            AccU acc; acc.v = zero;
            #pragma unroll
            for (int kk = 0; kk < 2; kk++) {
                FragBF bk_;
                int n = cb * 16 + l15;
                const unsigned* p = Kw + n * 32 + kk * 16 + hf * 4;
                #pragma unroll
                for (int j = 0; j < 4; j++) { bk_.u[j] = p[j]; bk_.u[4 + j] = p[8 + j]; }
                acc.v = wmma_bf16(aq[kk], bk_, acc.v);
            }
            #pragma unroll
            for (int vi = 0; vi < 8; vi++)
                Sw[(vi + 8 * hf) * 256 + cb * 16 + l15] = acc.f[vi] * scale;
        }
        // softmax over 256 keys: lane handles row l15, half of the columns
        {
            float* Srow = Sw + l15 * 256 + hf * 128;
            float mx = -3.4e38f;
            for (int j = 0; j < 128; j++) mx = fmaxf(mx, Srow[j]);
            mx = fmaxf(mx, __shfl_xor(mx, 16));
            float sum = 0.f;
            for (int j = 0; j < 128; j++) { float e = __expf(Srow[j] - mx); Srow[j] = e; sum += e; }
            sum += __shfl_xor(sum, 16);
            float inv = 1.f / sum;
            for (int j = 0; j < 128; j++) Srow[j] *= inv;
        }
        // O = P V  (P fp32 in LDS -> bf16 fragments on the fly)
        for (int ob = 0; ob < 4; ob++) {
            AccU acc; acc.v = zero;
            for (int kb = 0; kb < 8; kb++) {
                FragBF ap, bv;
                const float* Sr = Sw + l15 * 256 + kb * 32;
                #pragma unroll
                for (int j = 0; j < 8; j++) {
                    int kk = (j < 4) ? (hf * 8 + 2 * j) : (16 + hf * 8 + 2 * (j - 4));
                    ap.u[j] = packbf(Sr[kk], Sr[kk + 1]);
                }
                int d = ob * 16 + l15;
                const unsigned* p = Vw + d * 128 + kb * 16 + hf * 4;
                #pragma unroll
                for (int j = 0; j < 4; j++) { bv.u[j] = p[j]; bv.u[4 + j] = p[8 + j]; }
                acc.v = wmma_bf16(ap, bv, acc.v);
            }
            #pragma unroll
            for (int vi = 0; vi < 8; vi++) {
                long gr = row0 + rb * 16 + vi + 8 * hf;
                o[gr * 512 + h * 64 + ob * 16 + l15] = acc.f[vi];
            }
        }
    }
}

// ---------------------------------------------------------------------------
// LayerNorm over last dim (one block per row)
// ---------------------------------------------------------------------------
__global__ __launch_bounds__(256) void ln_kernel(
        const float* __restrict__ x, const float* __restrict__ g,
        const float* __restrict__ b, float* __restrict__ y, int C) {
    __shared__ float red[256];
    long row = blockIdx.x;
    const float* xr = x + row * C;
    int t = threadIdx.x;
    float s = 0.f;
    for (int i = t; i < C; i += 256) s += xr[i];
    red[t] = s; __syncthreads();
    for (int o = 128; o > 0; o >>= 1) { if (t < o) red[t] += red[t + o]; __syncthreads(); }
    float mean = red[0] / C; __syncthreads();
    float v = 0.f;
    for (int i = t; i < C; i += 256) { float d = xr[i] - mean; v += d * d; }
    red[t] = v; __syncthreads();
    for (int o = 128; o > 0; o >>= 1) { if (t < o) red[t] += red[t + o]; __syncthreads(); }
    float inv = rsqrtf(red[0] / C + 1e-5f);
    for (int i = t; i < C; i += 256) y[row * C + i] = (xr[i] - mean) * inv * g[i] + b[i];
}

// ---------------------------------------------------------------------------
// Causal conv1d with arbitrary in/out strides (folds the NCH<->NLC transposes)
// w layout: [Cout][Cin][K]
// ---------------------------------------------------------------------------
__global__ __launch_bounds__(256) void conv_kernel(
        const float* __restrict__ in, const float* __restrict__ w,
        const float* __restrict__ bias, float* __restrict__ out,
        int Bn, int Cin, int Cout, int L, int Kw,
        long iSB, long iSC, long iSL, long oSB, long oSC, long oSL) {
    int idx = blockIdx.x * 256 + threadIdx.x;
    int tot = Bn * Cout * L;
    if (idx >= tot) return;
    int l  = idx % L;
    int co = (idx / L) % Cout;
    int bb = idx / (L * Cout);
    float acc = bias[co];
    for (int ci = 0; ci < Cin; ci++) {
        const float* wr = w + (long)(co * Cin + ci) * Kw;
        for (int k = 0; k < Kw; k++) {
            int pos = l - (Kw - 1) + k;
            if (pos >= 0)
                acc += in[(long)bb * iSB + (long)ci * iSC + (long)pos * iSL] * wr[k];
        }
    }
    out[(long)bb * oSB + (long)co * oSC + (long)l * oSL] = acc;
}

// ---------------------------------------------------------------------------
// Residual VQ: 8 codebooks of 1024x64; output = sum of selected codes
// ---------------------------------------------------------------------------
__global__ __launch_bounds__(256) void vq_kernel(
        const float* __restrict__ x, const float* __restrict__ cbs,
        float* __restrict__ q, int rows) {
    int r = blockIdx.x * 256 + threadIdx.x;
    if (r >= rows) return;
    float res[64], out[64];
    #pragma unroll
    for (int j = 0; j < 64; j++) { res[j] = x[(long)r * 64 + j]; out[j] = 0.f; }
    for (int qi = 0; qi < 8; qi++) {
        const float* cb = cbs + (long)qi * 1024 * 64;
        int best = 0; float bd = 3.4e38f;
        for (int c = 0; c < 1024; c++) {
            float d = 0.f;
            const float* cr = cb + (long)c * 64;
            #pragma unroll
            for (int j = 0; j < 64; j++) { float t = res[j] - cr[j]; d += t * t; }
            if (d < bd) { bd = d; best = c; }
        }
        const float* bc = cb + (long)best * 64;
        #pragma unroll
        for (int j = 0; j < 64; j++) { res[j] -= bc[j]; out[j] += bc[j]; }
    }
    #pragma unroll
    for (int j = 0; j < 64; j++) q[(long)r * 64 + j] = out[j];
}

// ---------------------------------------------------------------------------
// Host orchestration
// ---------------------------------------------------------------------------
extern "C" void kernel_launch(void* const* d_in, const int* in_sizes, int n_in,
                              void* d_out, int out_size, void* d_ws, size_t ws_size,
                              hipStream_t stream) {
    (void)in_sizes; (void)n_in; (void)out_size; (void)ws_size;
    auto F = [&](int i) { return (const float*)d_in[i]; };

    // ---- walk the flattened parameter dict (setup_inputs order) ----
    int idx = 0;
    const float* sig = F(idx++);
    int enc_pre_w = idx++, enc_pre_b = idx++;
    int enc_dw[4], enc_db[4], enc_attn[4];
    const int enc_depth[4] = {2, 2, 2, 4};
    const int enc_co[4]    = {128, 256, 512, 1024};
    for (int l = 0; l < 4; l++) {
        enc_dw[l] = idx++; enc_db[l] = idx++;
        enc_attn[l] = idx; idx += 10 * enc_depth[l];
    }
    int post_w = idx++, post_b = idx++, norm_g = idx++, norm_b = idx++;
    int cb_idx = idx++;
    int dec_pre_w = idx++, dec_pre_b = idx++;
    int dec_uw[4], dec_ub[4], dec_attn[4];
    const int dec_depth[4] = {4, 2, 2, 2};
    const int dec_co[4]    = {512, 256, 128, 128};
    for (int l = 0; l < 4; l++) {
        dec_uw[l] = idx++; dec_ub[l] = idx++;
        dec_attn[l] = idx; idx += 10 * dec_depth[l];
    }
    int lng = idx++, lnb = idx++, outw = idx++, outb = idx++;

    // ---- scratch partition (bump allocator over d_ws) ----
    char* wp = (char*)d_ws;
    auto alloc = [&](size_t nf) -> float* {
        float* p = (float*)wp;
        wp += ((nf * sizeof(float) + 255) & ~(size_t)255);
        return p;
    };
    float* X   = alloc(4ull * 1024 * 1024 + 131072);   // activations ping
    float* Yb  = alloc(4ull * 1024 * 1024 + 131072);   // activations pong
    float* H   = alloc(4ull * 1024 * 1024 + 131072);   // layernorm output
    float* QKV = alloc(50ull * 1024 * 1024 + 1048576); // 32768 x 1536 max
    float* O   = alloc(17ull * 1024 * 1024);           // 32768 x 512 max
    float* FF  = alloc(17ull * 1024 * 1024);           // 32768 x 512/4096 max

    auto gemm = [&](const float* A, const float* W, const float* bias,
                    const float* res, float* Yo, int M, int N, int K, int flags) {
        dim3 g(N / 64, M / 128);
        gemm_kernel<<<g, 256, 0, stream>>>(A, W, bias, res, Yo, M, N, K, flags);
    };
    auto attn_stack = [&](float* x, int M, int dim, int base, int depth) {
        for (int d = 0; d < depth; d++) {
            int p = base + d * 10;  // ln1_g,ln1_b,wqkv,wo,ln2_g,ln2_b,w1,b1,w2,b2
            ln_kernel<<<M, 256, 0, stream>>>(x, F(p + 0), F(p + 1), H, dim);
            gemm(H, F(p + 2), nullptr, nullptr, QKV, M, 1536, dim, 0);
            attn_kernel<<<(M / 256) * 8, 256, ATTN_SMEM, stream>>>(QKV, O);
            gemm(O, F(p + 3), nullptr, x, x, M, dim, 512, FLAG_RES);
            ln_kernel<<<M, 256, 0, stream>>>(x, F(p + 4), F(p + 5), H, dim);
            gemm(H, F(p + 6), F(p + 7), nullptr, FF, M, 4 * dim, dim, FLAG_BIAS | FLAG_GELU);
            gemm(FF, F(p + 8), F(p + 9), x, x, M, dim, 4 * dim, FLAG_BIAS | FLAG_RES);
        }
    };
    auto conv = [&](const float* in, int wi, int bi, float* out,
                    int Bn, int Cin, int Cout, int L, int Kw,
                    long iSB, long iSC, long iSL, long oSB, long oSC, long oSL) {
        long tot = (long)Bn * Cout * L;
        conv_kernel<<<(int)((tot + 255) / 256), 256, 0, stream>>>(
            in, F(wi), F(bi), out, Bn, Cin, Cout, L, Kw,
            iSB, iSC, iSL, oSB, oSC, oSL);
    };

    const int B = 2;
    int L = 16384;

    // ---- encoder ----
    // pre conv: sig (B,2,L) -> X as (B,L,128)
    conv(sig, enc_pre_w, enc_pre_b, X, B, 2, 128, L, 7,
         (long)2 * L, L, 1, (long)L * 128, 1, 128);
    int C = 128;
    float* cur = X; float* nxt = Yb;
    for (int l = 0; l < 4; l++) {
        int co = enc_co[l];
        int Mrows = B * (L / 2);   // (B, L/2, 2C) view is contiguous
        gemm(cur, F(enc_dw[l]), F(enc_db[l]), nullptr, nxt, Mrows, co, 2 * C, FLAG_BIAS);
        { float* tswap = cur; cur = nxt; nxt = tswap; }
        L /= 2; C = co;
        attn_stack(cur, B * L, C, enc_attn[l], enc_depth[l]);
    }
    // post conv: cur (B,L,1024) channels-last -> nxt as (B,L,64)
    conv(cur, post_w, post_b, nxt, B, 1024, 64, L, 3,
         (long)L * 1024, 1, 1024, (long)L * 64, 1, 64);
    // encoder LN -> cur (reused as scratch)
    ln_kernel<<<B * L, 256, 0, stream>>>(nxt, F(norm_g), F(norm_b), cur, 64);
    // residual VQ -> nxt
    vq_kernel<<<(B * L + 255) / 256, 256, 0, stream>>>(cur, F(cb_idx), nxt, B * L);

    // ---- decoder ----
    // pre conv: nxt (B,L,64) -> cur as (B,L,1024)
    conv(nxt, dec_pre_w, dec_pre_b, cur, B, 64, 1024, L, 7,
         (long)L * 64, 1, 64, (long)L * 1024, 1, 1024);
    C = 1024;
    for (int l = 0; l < 4; l++) {
        int co = dec_co[l];
        gemm(cur, F(dec_uw[l]), F(dec_ub[l]), nullptr, nxt, B * L, 2 * co, C, FLAG_BIAS);
        { float* tswap = cur; cur = nxt; nxt = tswap; }
        L *= 2; C = co;                // (B,L,2,co) -> (B,2L,co) contiguous
        attn_stack(cur, B * L, C, dec_attn[l], dec_depth[l]);
    }
    // final LN -> H, then out conv: H (B,L,128) -> d_out as (B,2,L)
    ln_kernel<<<B * L, 256, 0, stream>>>(cur, F(lng), F(lnb), H, 128);
    conv(H, outw, outb, (float*)d_out, B, 128, 2, L, 3,
         (long)L * 128, 1, 128, (long)2 * L, L, 1);
}
